// GSTOPR_68813966016626
// MI455X (gfx1250) — matrix operand order repaired and checked by reading, compiled, not run on hardware
//
#include <hip/hip_runtime.h>
#include <hip/hip_bf16.h>

#define B_    512
#define L_    1024
#define NPG_  256
#define EMB_  128
#define E_    (B_ * L_)        // 524288 edges
#define K2_   256              // 2*EMB
#define N4_   512              // 4*EMB
#define MTILE 64               // edges per block
#define LDS_STRIDE 264         // 256 + 8 pad (f16) -> 132 dwords (== 4 mod 64, conflict-free)

typedef _Float16 v16h __attribute__((ext_vector_type(16)));
typedef _Float16 v8h  __attribute__((ext_vector_type(8)));
typedef _Float16 v4h  __attribute__((ext_vector_type(4)));
typedef float    v8f  __attribute__((ext_vector_type(8)));
typedef float    v4f  __attribute__((ext_vector_type(4)));

// ---------------------------------------------------------------------------
// Kernel 0: repack W1 (256x512 f32 row-major) into WMMA B-fragment layout f16.
// Fragment (nt, kt): 32 lanes x 16 f16. Lane l holds col n = nt*16 + (l&15),
// K = kt*32 + (l<16 ? 0 : 16) + i for i = 0..15  (contiguous k order, same as the A fragments).
// ---------------------------------------------------------------------------
__global__ __launch_bounds__(256) void GST_prep_w1(const float* __restrict__ W1,
                                                   _Float16* __restrict__ W1h) {
    int t    = blockIdx.x * 256 + threadIdx.x;   // 0..8191
    int lane = t & 31;
    int frag = t >> 5;                           // 0..255  == nt*8 + kt
    int kt   = frag & 7;
    int nt   = frag >> 3;
    int n    = nt * 16 + (lane & 15);
    int kb   = kt * 32 + ((lane & 16) ? 16 : 0);
    v16h v;
#pragma unroll
    for (int i = 0; i < 16; ++i)
        v[i] = (_Float16)W1[(kb + i) * N4_ + n];
    *(v16h*)(W1h + frag * 512 + lane * 16) = v;
}

// ---------------------------------------------------------------------------
// Kernel 1: fused gather + GEMM(E x 256 @ 256 x 512) + bias + ReLU + @W2 + b2.
// Block = 64 threads (2 waves), 64 edges per block. Each wave owns 32 edge
// rows = TWO 16-row A-tiles resident in VGPRs, so every B-fragment load feeds
// two independent v_wmma issues (halves B traffic per edge vs 1 tile/wave).
// ---------------------------------------------------------------------------
__global__ __launch_bounds__(64) void GST_scores(const float* __restrict__ h,
                                                 const _Float16* __restrict__ W1h,
                                                 const float* __restrict__ b1,
                                                 const float* __restrict__ W2,
                                                 const float* __restrict__ b2,
                                                 const int* __restrict__ edge_local,
                                                 float* __restrict__ scores) {
    __shared__ __align__(16) _Float16 sA[MTILE * LDS_STRIDE];

    const int tid   = threadIdx.x;
    const int lane  = tid & 31;
    const int wave  = tid >> 5;                  // 0..1
    const int eBase = blockIdx.x * MTILE;
    const int offs  = (eBase >> 10) * NPG_;      // graph id * NPG (64 | 1024)

    // ---- Phase 1: gather h rows -> LDS (f32 -> f16), edge_rep tile [64][256]
    // 128 half-row tasks: task t -> edge t>>1, half t&1; 32 lanes x 4 floats each.
#pragma unroll 4
    for (int it = 0; it < 64; ++it) {
        int task = it * 2 + wave;
        int edge = task >> 1;
        int half = task & 1;
        int node = edge_local[half * E_ + (eBase + edge)] + offs;
        v4f v = *(const v4f*)(h + node * EMB_ + lane * 4);
        v4h hv;
#pragma unroll
        for (int i = 0; i < 4; ++i) hv[i] = (_Float16)v[i];
        *(v4h*)(&sA[edge * LDS_STRIDE + half * EMB_ + lane * 4]) = hv;
    }
    __syncthreads();

    // ---- Phase 2: load all A-fragments (K = 256) for this wave's 32 rows.
    // A layout (16x32 f16), matching B: lanes 0-15 K=[0,16); lanes 16-31 K=[16,32).
    const int kShift = (lane < 16) ? 0 : 16;
    const _Float16* aRow0 = &sA[(wave * 32 + (lane & 15)) * LDS_STRIDE + kShift];
    const _Float16* aRow1 = aRow0 + 16 * LDS_STRIDE;
    v16h afrag0[8], afrag1[8];
#pragma unroll
    for (int kt = 0; kt < 8; ++kt) {
        v8h lo0 = *(const v8h*)(aRow0 + kt * 32);        // ds_load_b128, conflict-free
        v8h hi0 = *(const v8h*)(aRow0 + kt * 32 + 8);
        afrag0[kt] = __builtin_shufflevector(lo0, hi0, 0, 1, 2, 3, 4, 5, 6, 7,
                                             8, 9, 10, 11, 12, 13, 14, 15);
        v8h lo1 = *(const v8h*)(aRow1 + kt * 32);
        v8h hi1 = *(const v8h*)(aRow1 + kt * 32 + 8);
        afrag1[kt] = __builtin_shufflevector(lo1, hi1, 0, 1, 2, 3, 4, 5, 6, 7,
                                             8, 9, 10, 11, 12, 13, 14, 15);
    }

    const int colN = lane & 15;
    float score0[8] = {0.f, 0.f, 0.f, 0.f, 0.f, 0.f, 0.f, 0.f};
    float score1[8] = {0.f, 0.f, 0.f, 0.f, 0.f, 0.f, 0.f, 0.f};

    for (int nt = 0; nt < 32; ++nt) {
        const _Float16* bBase = W1h + (nt * 8) * 512 + lane * 16;
        // Pull next nt's 8KB B-slab toward L0 while this one computes.
        if (nt < 31) __builtin_prefetch(bBase + 8 * 512, 0, 1);
        v8f c0 = {0.f, 0.f, 0.f, 0.f, 0.f, 0.f, 0.f, 0.f};
        v8f c1 = {0.f, 0.f, 0.f, 0.f, 0.f, 0.f, 0.f, 0.f};
#pragma unroll
        for (int kt = 0; kt < 8; ++kt) {
            v16h bfrag = *(const v16h*)(bBase + kt * 512);
            // Two independent WMMAs per B fragment (different accumulators).
            c0 = __builtin_amdgcn_wmma_f32_16x16x32_f16(
                     false, afrag0[kt], false, bfrag, (short)0, c0, false, false);
            c1 = __builtin_amdgcn_wmma_f32_16x16x32_f16(
                     false, afrag1[kt], false, bfrag, (short)0, c1, false, false);
        }
        // Fused epilogue: +b1, ReLU, dot with W2 column weights.
        float bias = b1[nt * 16 + colN];
        float w2v  = W2[nt * 16 + colN];
#pragma unroll
        for (int j = 0; j < 8; ++j) {
            float r0 = c0[j] + bias;
            r0 = r0 > 0.f ? r0 : 0.f;
            score0[j] += r0 * w2v;
            float r1 = c1[j] + bias;
            r1 = r1 > 0.f ? r1 : 0.f;
            score1[j] += r1 * w2v;
        }
    }

    // Reduce over the 16 lanes holding the 16 columns of each C tile half.
#pragma unroll
    for (int j = 0; j < 8; ++j) {
        float s0 = score0[j];
        s0 += __shfl_xor(s0, 1);
        s0 += __shfl_xor(s0, 2);
        s0 += __shfl_xor(s0, 4);
        s0 += __shfl_xor(s0, 8);
        score0[j] = s0;
        float s1 = score1[j];
        s1 += __shfl_xor(s1, 1);
        s1 += __shfl_xor(s1, 2);
        s1 += __shfl_xor(s1, 4);
        s1 += __shfl_xor(s1, 8);
        score1[j] = s1;
    }
    if ((lane & 15) == 0) {
        float bb    = b2[0];
        int   rbase = eBase + wave * 32 + ((lane < 16) ? 0 : 8);
#pragma unroll
        for (int j = 0; j < 8; ++j) {
            scores[rbase + j]      = score0[j] + bb;
            scores[rbase + 16 + j] = score1[j] + bb;
        }
    }
}

// ---------------------------------------------------------------------------
// Kernel 2: per-graph standardization + 30-iter 2-column Sinkhorn + exp.
// One block (256 threads, 8 waves) per graph; 4 edges per thread in registers.
// ---------------------------------------------------------------------------
__device__ __forceinline__ float GST_blockReduce(float v, float* red, bool isMax) {
#pragma unroll
    for (int m = 16; m >= 1; m >>= 1) {
        float o = __shfl_xor(v, m);
        v = isMax ? fmaxf(v, o) : (v + o);
    }
    __syncthreads();                       // WAR protection for 'red'
    if ((threadIdx.x & 31) == 0) red[threadIdx.x >> 5] = v;
    __syncthreads();
    float r = red[0];
#pragma unroll
    for (int k = 1; k < 8; ++k) r = isMax ? fmaxf(r, red[k]) : (r + red[k]);
    return r;
}

__global__ __launch_bounds__(256) void GST_sinkhorn(const float* __restrict__ scores,
                                                    float* __restrict__ out) {
    __shared__ float red[8];
    const int tid = threadIdx.x;
    const int b   = blockIdx.x;

    float a[4];
#pragma unroll
    for (int i = 0; i < 4; ++i) a[i] = scores[b * L_ + i * 256 + tid];

    // mean / unbiased std (ddof = 1)
    float mu = GST_blockReduce(a[0] + a[1] + a[2] + a[3], red, false) * (1.f / 1024.f);
    float sq = 0.f;
#pragma unroll
    for (int i = 0; i < 4; ++i) { float d = a[i] - mu; sq += d * d; }
    float var    = GST_blockReduce(sq, red, false) * (1.f / 1023.f);
    float inv_sd = 1.f / sqrtf(var);

    float lt0[4], lt1[4];
#pragma unroll
    for (int i = 0; i < 4; ++i) {
        float att = (a[i] - mu) * inv_sd;
        lt0[i] = -att;          // -D[...,0]
        lt1[i] = att - 1.f;     // -D[...,1]
    }

    const float LR0 = 6.643789733f;   // log(L - 0.25*L + eps) = log(768)
    const float LR1 = 5.545177444f;   // log(0.25*L + eps)     = log(256)

    for (int it = 0; it < 30; ++it) {
        float m0 = -1e30f, m1 = -1e30f;
#pragma unroll
        for (int i = 0; i < 4; ++i) {
            // row logsumexp over the 2 columns
            float m   = fmaxf(lt0[i], lt1[i]);
            float lse = m + logf(expf(lt0[i] - m) + expf(lt1[i] - m));
            lt0[i] -= lse;  lt1[i] -= lse;
            m0 = fmaxf(m0, lt0[i]);  m1 = fmaxf(m1, lt1[i]);
        }
        // column logsumexp over L = 1024 edges
        m0 = GST_blockReduce(m0, red, true);
        m1 = GST_blockReduce(m1, red, true);
        float s0 = 0.f, s1 = 0.f;
#pragma unroll
        for (int i = 0; i < 4; ++i) {
            s0 += expf(lt0[i] - m0);
            s1 += expf(lt1[i] - m1);
        }
        s0 = GST_blockReduce(s0, red, false);
        s1 = GST_blockReduce(s1, red, false);
        float add0 = LR0 - (m0 + logf(s0));
        float add1 = LR1 - (m1 + logf(s1));
#pragma unroll
        for (int i = 0; i < 4; ++i) { lt0[i] += add0; lt1[i] += add1; }
    }

#pragma unroll
    for (int i = 0; i < 4; ++i)
        out[b * L_ + i * 256 + tid] = expf(lt1[i]);
}

// ---------------------------------------------------------------------------
extern "C" void kernel_launch(void* const* d_in, const int* in_sizes, int n_in,
                              void* d_out, int out_size, void* d_ws, size_t ws_size,
                              hipStream_t stream) {
    const float* h    = (const float*)d_in[0];
    const float* W1   = (const float*)d_in[1];
    const float* b1   = (const float*)d_in[2];
    const float* W2   = (const float*)d_in[3];
    const float* b2   = (const float*)d_in[4];
    const int* edge_local = (const int*)d_in[5];
    float* out = (float*)d_out;

    _Float16* W1h    = (_Float16*)d_ws;                       // 256 KB
    float*    scores = (float*)((char*)d_ws + 256 * 1024);    // 2 MB

    GST_prep_w1 <<<32,   256, 0, stream>>>(W1, W1h);
    GST_scores  <<<E_ / MTILE, 64, 0, stream>>>(h, W1h, b1, W2, b2, edge_local, scores);
    GST_sinkhorn<<<B_,   256, 0, stream>>>(scores, out);
}
